// TransformerBlock_20650202759205
// MI455X (gfx1250) — matrix operand order repaired and checked by reading, not compiled, not run on hardware
//
#include <hip/hip_runtime.h>

// ---------------------------------------------------------------------------
// Transformer block (pre-norm MHA + pre-norm ReGLU FFN) for MI455X / gfx1250.
// All GEMMs run through v_wmma_f32_16x16x32_bf16 (fp32 accumulate).
// GEMM weight tiles are staged in LDS (double-buffered); on toolchains that
// expose them, staging uses GLOBAL_LOAD_ASYNC_TO_LDS_B128 + s_wait_asynccnt.
// Attention is flash-style (online softmax), so no S x S score buffer.
// ---------------------------------------------------------------------------

#if defined(__has_builtin)
#if __has_builtin(__builtin_amdgcn_global_load_async_to_lds_b128) && \
    __has_builtin(__builtin_amdgcn_s_wait_asynccnt)
#define USE_ASYNC_LDS 1
#endif
#endif
#ifndef USE_ASYNC_LDS
#define USE_ASYNC_LDS 0
#endif

namespace {

constexpr int kD  = 1024;           // d_model
constexpr int kS  = 2048;           // sequence length
constexpr int kB  = 4;              // batch
constexpr int kT  = kB * kS;        // 8192 tokens
constexpr int kH  = 16;             // heads
constexpr int kDk = 64;             // head dim
constexpr int kF  = 2048;           // FFN hidden (per ReGLU half)

// LDS B-tile row pitch: 40 bf16 = 80 B = 20 DWORD banks -> 16 distinct banks
// across the 16 fragment lanes, and 80 B is 16B-aligned for ds_load_b128.
constexpr int kBP = 40;

typedef __attribute__((ext_vector_type(16))) __bf16 v16bf;
typedef __attribute__((ext_vector_type(8)))  __bf16 v8bf;
typedef __attribute__((ext_vector_type(8)))  float  v8f;
typedef __attribute__((ext_vector_type(4)))  int    v4i;

#if USE_ASYNC_LDS
typedef __attribute__((address_space(1))) v4i* gv4i_p;
typedef __attribute__((address_space(3))) v4i* lv4i_p;
#endif

__device__ inline v8f zero8() {
  v8f z;
#pragma unroll
  for (int i = 0; i < 8; ++i) z[i] = 0.0f;
  return z;
}

// 16B global->LDS copy: async (ASYNCcnt) when available, sync fallback.
__device__ inline void copy16B_to_lds(const __bf16* g, __bf16* l) {
#if USE_ASYNC_LDS
  __builtin_amdgcn_global_load_async_to_lds_b128((gv4i_p)g, (lv4i_p)l, 0, 0);
#else
  *(v4i*)l = *(const v4i*)g;
#endif
}

__device__ inline void async_join() {
#if USE_ASYNC_LDS
  __builtin_amdgcn_s_wait_asynccnt(0);
#endif
}

// A-fragment (16x32, lane = M row). Per ISA layout: lanes 0-15 hold K
// {0..7, 16..23}; lanes 16-31 hold K {8..15, 24..31}. Two 16B loads.
__device__ inline v16bf load_frag_a(const __bf16* base, int ld) {
  const int lane = threadIdx.x & 31;
  const __bf16* p = base + (size_t)(lane & 15) * ld + ((lane >> 4) << 3);
  v8bf lo = *(const v8bf*)(p);
  v8bf hi = *(const v8bf*)(p + 16);
  v16bf r;
#pragma unroll
  for (int i = 0; i < 8; ++i) { r[i] = lo[i]; r[i + 8] = hi[i]; }
  return r;
}

// B-fragment (32x16, lane = N column; column data = contiguous row of the
// [N,K] weight matrix). Lanes 0-15 hold K {0..7, 16..23}; lanes 16-31 hold K {8..15, 24..31}.
__device__ inline v16bf load_frag_b(const __bf16* base, int ld) {
  const int lane = threadIdx.x & 31;
  const __bf16* p = base + (size_t)(lane & 15) * ld + ((lane >> 4) << 3);
  v8bf lo = *(const v8bf*)(p);
  v8bf hi = *(const v8bf*)(p + 16);
  v16bf r;
#pragma unroll
  for (int i = 0; i < 8; ++i) { r[i] = lo[i]; r[i + 8] = hi[i]; }
  return r;
}

__device__ inline v8f wmma_bf16(v16bf a, v16bf b, v8f c) {
  // (neg_a, A, neg_b, B, c_mod, C, reuse_a, reuse_b)
  return __builtin_amdgcn_wmma_f32_16x16x32_bf16(false, a, false, b,
                                                 (short)0, c, false, false);
}

// ---------------------------------------------------------------------------
__global__ void cvt_f32_bf16(const float* __restrict__ in,
                             __bf16* __restrict__ out, int n) {
  for (int i = blockIdx.x * blockDim.x + threadIdx.x; i < n;
       i += gridDim.x * blockDim.x)
    out[i] = (__bf16)in[i];
}

// ---------------------------------------------------------------------------
// LayerNorm over rows of 1024 fp32, emits bf16. One 256-thread block per row.
__global__ __launch_bounds__(256)
void layernorm_bf16(const float* __restrict__ x, const float* __restrict__ gamma,
                    const float* __restrict__ beta, __bf16* __restrict__ out) {
  __shared__ float s1[256];
  __shared__ float s2[256];
  const int row = blockIdx.x, tid = threadIdx.x;
  const float4 v = ((const float4*)(x + (size_t)row * kD))[tid];
  s1[tid] = v.x + v.y + v.z + v.w;
  s2[tid] = v.x * v.x + v.y * v.y + v.z * v.z + v.w * v.w;
  __syncthreads();
  for (int off = 128; off > 0; off >>= 1) {
    if (tid < off) { s1[tid] += s1[tid + off]; s2[tid] += s2[tid + off]; }
    __syncthreads();
  }
  const float mu   = s1[0] * (1.0f / kD);
  const float var  = s2[0] * (1.0f / kD) - mu * mu;
  const float rstd = rsqrtf(var + 1e-5f);
  const int c = tid * 4;
  __bf16* o = out + (size_t)row * kD + c;
  o[0] = (__bf16)((v.x - mu) * rstd * gamma[c + 0] + beta[c + 0]);
  o[1] = (__bf16)((v.y - mu) * rstd * gamma[c + 1] + beta[c + 1]);
  o[2] = (__bf16)((v.z - mu) * rstd * gamma[c + 2] + beta[c + 2]);
  o[3] = (__bf16)((v.w - mu) * rstd * gamma[c + 3] + beta[c + 3]);
}

// ---------------------------------------------------------------------------
// C[M,N] = A[M,K] * W[N,K]^T (+ epilogue). 256 threads = 8 waves; block tile
// 64x128; each wave owns a 16x64 strip (4 f32 accumulators). The 128x32
// weight tile is staged in LDS (double-buffered, async copy when available)
// and shared by the 4 row-waves; A fragments stream straight from L2.
//  MODE 0: out_bf16 = C + bias
//  MODE 1: out_f32  = C + bias + R        (residual add)
//  MODE 2: out_bf16 = relu(C_a+bias_a) * (C_b+bias_b)   (ReGLU, two weights)
template <int MODE>
__global__ __launch_bounds__(256)
void gemm_wmma(const __bf16* __restrict__ A,
               const __bf16* __restrict__ W, const __bf16* __restrict__ W2,
               const float* __restrict__ bias, const float* __restrict__ bias2,
               const float* __restrict__ R,
               float* __restrict__ outF, __bf16* __restrict__ outB,
               int N, int K) {
  __shared__ __bf16 ldsB[2][128 * kBP];
  __shared__ __bf16 ldsB2[(MODE == 2) ? 2 : 1][(MODE == 2) ? 128 * kBP : 1];

  const int wave = threadIdx.x >> 5;
  const int lane = threadIdx.x & 31;
  const int m0 = blockIdx.x * 64 + (wave & 3) * 16;
  const int cw = wave >> 2;                 // column half within the block
  const int nblk = blockIdx.y * 128;        // first output column of block

  v8f acc[4], accb[4];
#pragma unroll
  for (int t = 0; t < 4; ++t) { acc[t] = zero8(); accb[t] = zero8(); }

  const __bf16* Arow = A + (size_t)m0 * K;
  const __bf16* Wblk = W + (size_t)nblk * K;
  const __bf16* W2blk = (MODE == 2) ? W2 + (size_t)nblk * K : nullptr;

  // Stage one 128x32 tile (512 x 16B chunks; 2 per thread) at column k0.
  auto stage = [&](const __bf16* src, __bf16* dst, int k0) {
    for (int c = threadIdx.x; c < 512; c += 256) {
      const int row = c >> 2, seg = c & 3;
      copy16B_to_lds(src + (size_t)row * K + k0 + seg * 8,
                     dst + row * kBP + seg * 8);
    }
  };

  stage(Wblk, ldsB[0], 0);
  if (MODE == 2) stage(W2blk, ldsB2[0], 0);
  async_join();
  __syncthreads();

  for (int k0 = 0; k0 < K; k0 += 32) {
    const int cur = (k0 >> 5) & 1, nxt = cur ^ 1;
    if (k0 + 32 < K) {                       // overlap next tile with compute
      stage(Wblk, ldsB[nxt], k0 + 32);
      if (MODE == 2) stage(W2blk, ldsB2[nxt], k0 + 32);
    }
    if (k0 + 64 < K) __builtin_prefetch(Arow + k0 + 64, 0, 0);  // global_prefetch_b8

    v16bf af = load_frag_a(Arow + k0, K);
#pragma unroll
    for (int t = 0; t < 4; ++t) {
      v16bf bf = load_frag_b(&ldsB[cur][(cw * 64 + t * 16) * kBP], kBP);
      acc[t] = wmma_bf16(af, bf, acc[t]);
      if (MODE == 2) {
        v16bf bg = load_frag_b(&ldsB2[cur][(cw * 64 + t * 16) * kBP], kBP);
        accb[t] = wmma_bf16(af, bg, accb[t]);
      }
    }
    async_join();      // next tile fully in LDS (this wave's copies)
    __syncthreads();   // ...and everyone done reading `cur` before reuse
  }

  // C layout: VGPR j -> M = j + 8*(lane>=16); N = lane & 15.
  const int nlane = lane & 15;
  const int mbase = (lane >> 4) * 8;
#pragma unroll
  for (int t = 0; t < 4; ++t) {
#pragma unroll
    for (int j = 0; j < 8; ++j) {
      const int m = m0 + mbase + j;
      const int n = nblk + cw * 64 + t * 16 + nlane;
      const size_t idx = (size_t)m * N + n;
      float v = acc[t][j] + bias[n];
      if (MODE == 0) {
        outB[idx] = (__bf16)v;
      } else if (MODE == 1) {
        outF[idx] = v + R[idx];
      } else {
        const float vb = accb[t][j] + bias2[n];
        outB[idx] = (__bf16)(fmaxf(v, 0.0f) * vb);
      }
    }
  }
}

// ---------------------------------------------------------------------------
// Flash attention. Grid (S/64, H, B); 128 threads = 4 waves; wave w owns the
// 16 query rows m0 = 64*blockIdx.x + 16*w. Streams 64-key tiles:
//   S = Q K^T / 8  (WMMA, both operands contiguous over K = Dk)
//   online softmax in fp32 (16-lane shfl_xor reductions within half-waves)
//   ctx += P V     (P via per-wave LDS strip; V transposed into LDS)
__global__ __launch_bounds__(128)
void attn_flash(const __bf16* __restrict__ Q, const __bf16* __restrict__ Kmat,
                const __bf16* __restrict__ V, __bf16* __restrict__ O) {
  __shared__ __bf16 ldsVT[64 * 80];      // V^T tile: [dk][key], stride 80
  __shared__ __bf16 ldsP[4 * 16 * 80];   // per-wave P strips: [16][64+pad]

  const int tid = threadIdx.x, wave = tid >> 5, lane = tid & 31;
  const int nlane = lane & 15;
  const int mbase = (lane >> 4) * 8;
  const size_t tok0 = (size_t)blockIdx.z * kS;
  const int hcol = blockIdx.y * kDk;
  const int m0 = blockIdx.x * 64 + wave * 16;

  v16bf qf[2];
#pragma unroll
  for (int kc = 0; kc < 2; ++kc)
    qf[kc] = load_frag_a(Q + (tok0 + m0) * kD + hcol + kc * 32, kD);

  v8f acc[4];
#pragma unroll
  for (int t = 0; t < 4; ++t) acc[t] = zero8();
  float mstat[8], lstat[8];
#pragma unroll
  for (int j = 0; j < 8; ++j) { mstat[j] = -1.0e30f; lstat[j] = 0.0f; }

  for (int kv = 0; kv < kS; kv += 64) {
    __syncthreads();  // protect ldsVT from previous iteration's readers
    // stage V tile transposed: ldsVT[d * 80 + k]
    for (int c = tid; c < 512; c += 128) {
      const int kk = c >> 3, dbase = (c & 7) * 8;
      v8bf vv = *(const v8bf*)(V + (tok0 + kv + kk) * kD + hcol + dbase);
#pragma unroll
      for (int e = 0; e < 8; ++e) ldsVT[(dbase + e) * 80 + kk] = vv[e];
    }
    __syncthreads();

    // scores (16 queries x 64 keys), scaled by 1/sqrt(Dk) = 0.125
    float s[4][8];
#pragma unroll
    for (int nt = 0; nt < 4; ++nt) {
      v8f sc = zero8();
#pragma unroll
      for (int kc = 0; kc < 2; ++kc)
        sc = wmma_bf16(qf[kc],
                       load_frag_b(Kmat + (tok0 + kv + nt * 16) * kD + hcol + kc * 32, kD),
                       sc);
#pragma unroll
      for (int j = 0; j < 8; ++j) s[nt][j] = sc[j] * 0.125f;
    }

    // online softmax per query row (row spread over 16 lanes of a half-wave)
#pragma unroll
    for (int j = 0; j < 8; ++j) {
      float tmax = fmaxf(fmaxf(s[0][j], s[1][j]), fmaxf(s[2][j], s[3][j]));
#pragma unroll
      for (int off = 1; off < 16; off <<= 1)
        tmax = fmaxf(tmax, __shfl_xor(tmax, off, 32));
      const float mn = fmaxf(mstat[j], tmax);
      const float scl = __expf(mstat[j] - mn);
      mstat[j] = mn;
      float ps = 0.0f;
#pragma unroll
      for (int nt = 0; nt < 4; ++nt) {
        const float p = __expf(s[nt][j] - mn);
        s[nt][j] = p;
        ps += p;
      }
#pragma unroll
      for (int off = 1; off < 16; off <<= 1) ps += __shfl_xor(ps, off, 32);
      lstat[j] = lstat[j] * scl + ps;
#pragma unroll
      for (int nt = 0; nt < 4; ++nt) acc[nt][j] *= scl;
    }

    // P -> wave-private LDS strip (C-frag layout -> row major [m][key])
    __bf16* pw = ldsP + wave * 16 * 80;
#pragma unroll
    for (int nt = 0; nt < 4; ++nt)
#pragma unroll
      for (int j = 0; j < 8; ++j)
        pw[(mbase + j) * 80 + nt * 16 + nlane] = (__bf16)s[nt][j];

    // ctx += P (16x64) * V (64x64): A-frags from ldsP, B-frags from ldsVT
#pragma unroll
    for (int nt = 0; nt < 4; ++nt) {
#pragma unroll
      for (int kc = 0; kc < 2; ++kc) {
        v16bf pf = load_frag_a(pw + kc * 32, 80);
        v16bf vf = load_frag_b(ldsVT + (size_t)nt * 16 * 80 + kc * 32, 80);
        acc[nt] = wmma_bf16(pf, vf, acc[nt]);
      }
    }
  }

#pragma unroll
  for (int j = 0; j < 8; ++j) lstat[j] = 1.0f / lstat[j];
#pragma unroll
  for (int nt = 0; nt < 4; ++nt)
#pragma unroll
    for (int j = 0; j < 8; ++j)
      O[(tok0 + m0 + mbase + j) * kD + hcol + nt * 16 + nlane] =
          (__bf16)(acc[nt][j] * lstat[j]);
}

}  // namespace

// ---------------------------------------------------------------------------
extern "C" void kernel_launch(void* const* d_in, const int* in_sizes, int n_in,
                              void* d_out, int out_size, void* d_ws, size_t ws_size,
                              hipStream_t stream) {
  (void)in_sizes; (void)n_in; (void)out_size; (void)ws_size;

  const float* x   = (const float*)d_in[0];
  const float* Wq  = (const float*)d_in[1];
  const float* bq  = (const float*)d_in[2];
  const float* Wk  = (const float*)d_in[3];
  const float* bk  = (const float*)d_in[4];
  const float* Wv  = (const float*)d_in[5];
  const float* bv  = (const float*)d_in[6];
  const float* Wo  = (const float*)d_in[7];
  const float* bo  = (const float*)d_in[8];
  const float* g1  = (const float*)d_in[9];
  const float* be1 = (const float*)d_in[10];
  const float* g2  = (const float*)d_in[11];
  const float* be2 = (const float*)d_in[12];
  const float* W1  = (const float*)d_in[13];
  const float* b1  = (const float*)d_in[14];
  const float* W2  = (const float*)d_in[15];
  const float* b2  = (const float*)d_in[16];
  float* out = (float*)d_out;

  // Workspace layout (bytes). Total: 132 MB.
  char* ws = (char*)d_ws;
  const size_t MB = 1024u * 1024u;
  __bf16* wqb = (__bf16*)(ws + 0 * MB);    //  2 MB
  __bf16* wkb = (__bf16*)(ws + 2 * MB);    //  2 MB
  __bf16* wvb = (__bf16*)(ws + 4 * MB);    //  2 MB
  __bf16* wob = (__bf16*)(ws + 6 * MB);    //  2 MB
  __bf16* w1b = (__bf16*)(ws + 8 * MB);    //  8 MB
  __bf16* w2b = (__bf16*)(ws + 16 * MB);   //  4 MB
  __bf16* xn  = (__bf16*)(ws + 20 * MB);   // 16 MB (xn1, reused for xn2)
  __bf16* Qb  = (__bf16*)(ws + 36 * MB);   // 16 MB
  __bf16* Kb  = (__bf16*)(ws + 52 * MB);   // 16 MB
  __bf16* Vb  = (__bf16*)(ws + 68 * MB);   // 16 MB
  __bf16* Cb  = (__bf16*)(ws + 84 * MB);   // 16 MB (attention context)
  float*  o1  = (float*)(ws + 100 * MB);   // 32 MB (x + attn residual, fp32)
  __bf16* Gb  = (__bf16*)(ws + 36 * MB);   // 32 MB ReGLU out, aliases dead Q+K

  // 1) weights fp32 -> bf16 (L2-resident afterwards)
  cvt_f32_bf16<<<2048, 256, 0, stream>>>(Wq, wqb, kD * kD);
  cvt_f32_bf16<<<2048, 256, 0, stream>>>(Wk, wkb, kD * kD);
  cvt_f32_bf16<<<2048, 256, 0, stream>>>(Wv, wvb, kD * kD);
  cvt_f32_bf16<<<2048, 256, 0, stream>>>(Wo, wob, kD * kD);
  cvt_f32_bf16<<<2048, 256, 0, stream>>>(W1, w1b, 2 * kF * kD);
  cvt_f32_bf16<<<2048, 256, 0, stream>>>(W2, w2b, kD * kF);

  // 2) ln1(x) -> xn (bf16)
  layernorm_bf16<<<kT, 256, 0, stream>>>(x, g1, be1, xn);

  // 3) Q/K/V projections (bf16 out)
  dim3 gemmGrid(kT / 64, kD / 128);
  gemm_wmma<0><<<gemmGrid, 256, 0, stream>>>(xn, wqb, nullptr, bq, nullptr,
                                             nullptr, nullptr, Qb, kD, kD);
  gemm_wmma<0><<<gemmGrid, 256, 0, stream>>>(xn, wkb, nullptr, bk, nullptr,
                                             nullptr, nullptr, Kb, kD, kD);
  gemm_wmma<0><<<gemmGrid, 256, 0, stream>>>(xn, wvb, nullptr, bv, nullptr,
                                             nullptr, nullptr, Vb, kD, kD);

  // 4) flash attention -> ctx (bf16)
  attn_flash<<<dim3(kS / 64, kH, kB), 128, 0, stream>>>(Qb, Kb, Vb, Cb);

  // 5) o1 = x + ctx @ Wo^T + bo   (fp32)
  gemm_wmma<1><<<gemmGrid, 256, 0, stream>>>(Cb, wob, nullptr, bo, nullptr,
                                             x, o1, nullptr, kD, kD);

  // 6) ln2(o1) -> xn (bf16)
  layernorm_bf16<<<kT, 256, 0, stream>>>(o1, g2, be2, xn);

  // 7) ReGLU FFN up: g = relu(xn @ W1a^T + b1a) * (xn @ W1b^T + b1b)
  gemm_wmma<2><<<dim3(kT / 64, kF / 128), 256, 0, stream>>>(
      xn, w1b, w1b + (size_t)kF * kD, b1, b1 + kF, nullptr, nullptr, Gb, kF, kD);

  // 8) out = o1 + g @ W2^T + b2   (fp32, final output)
  gemm_wmma<1><<<gemmGrid, 256, 0, stream>>>(Gb, w2b, nullptr, b2, nullptr,
                                             o1, out, nullptr, kD, kF);
}
